// AttentionModel_39848706572648
// MI455X (gfx1250) — matrix operand order repaired and checked by reading, not compiled, not run on hardware
//
#include <hip/hip_runtime.h>

typedef __bf16 bf16_t;
typedef __attribute__((ext_vector_type(16))) __bf16 v16bf;
typedef __attribute__((ext_vector_type(8)))  __bf16 v8bf;
typedef __attribute__((ext_vector_type(8)))  float  v8f;
typedef __attribute__((ext_vector_type(16))) float  v16f;
typedef __attribute__((ext_vector_type(4)))  int    v4i;
typedef __attribute__((ext_vector_type(8)))  int    v8i;
typedef __attribute__((ext_vector_type(4)))  unsigned int v4u;

#define NB 2
#define NS 2048
#define NE 1024
#define NH 16
#define DH 64
#define NM (NB*NS)   // 4096 rows for the projection GEMMs

__device__ __forceinline__ bf16_t f2bf(float x) { return (bf16_t)x; }

__device__ __forceinline__ v8f wmma_bf16(v16bf a, v16bf b, v8f c) {
  // (neg_a, A, neg_b, B, c_mod, C, reuse_a, reuse_b)
  return __builtin_amdgcn_wmma_f32_16x16x32_bf16(false, a, false, b, (short)0, c,
                                                 false, false);
}

// ---- CDNA5 data-mover paths ------------------------------------------------
typedef __attribute__((address_space(1))) v4i*  g4i_t;
typedef __attribute__((address_space(3))) v4i*  l4i_t;
typedef __attribute__((address_space(3))) char* lchar_t;

#if __has_builtin(__builtin_amdgcn_tensor_load_to_lds)
#define HAVE_TDM 1
#else
#define HAVE_TDM 0
#endif
#if __has_builtin(__builtin_amdgcn_global_load_async_to_lds_b128)
#define HAVE_ASYNC_LDS 1
#if __has_builtin(__builtin_amdgcn_s_wait_asynccnt)
#define WAIT_ASYNCCNT(n) __builtin_amdgcn_s_wait_asynccnt(n)
#else
#define WAIT_ASYNCCNT(n) asm volatile("s_wait_asynccnt %0" :: "n"(n) : "memory")
#endif
#else
#define HAVE_ASYNC_LDS 0
#endif
#define HAVE_LDS_STAGE (HAVE_TDM || HAVE_ASYNC_LDS)

#if HAVE_TDM
#if __has_builtin(__builtin_amdgcn_s_wait_tensorcnt)
#define WAIT_TENSORCNT(n) __builtin_amdgcn_s_wait_tensorcnt(n)
#else
#define WAIT_TENSORCNT(n) asm volatile("s_wait_tensorcnt %0" :: "n"(n) : "memory")
#endif

// One TDM op copies a 16x32 f32 tile (2 KiB) from the row-major dropout
// tensor (row stride NS floats) into contiguous LDS. D# packing per ISA
// ch.8: group0 {count, lds_addr, global_addr, type=2}; group1 {data_size=4B,
// tensor_dim0/1, tile_dim0/1, tensor_dim0_stride}; groups 2-3 zero (2D tile).
// This toolchain exposes the 6-arg builtin (round-4 diagnostic).
__device__ __forceinline__ void tdm_load_dmask(const float* gsrc, float* lbuf) {
  unsigned long long ga = (unsigned long long)(size_t)gsrc;
  const unsigned int ga_lo   = __builtin_amdgcn_readfirstlane((unsigned int)ga);
  const unsigned int ga_hi   = __builtin_amdgcn_readfirstlane((unsigned int)(ga >> 32));
  const unsigned int lds_off = __builtin_amdgcn_readfirstlane(
      (unsigned int)(size_t)(lchar_t)lbuf);
  v4u g0;
  g0[0] = 1u;                                   // count=1 (valid user D#)
  g0[1] = lds_off;                              // lds_addr (bytes)
  g0[2] = ga_lo;                                // global_addr[31:0]
  g0[3] = (ga_hi & 0x01FFFFFFu) | (2u << 30);   // global_addr[56:32] | type=2
  v8i g1;
  g1[0] = (int)(2u << 16);                      // data_size=2 (4 bytes)
  g1[1] = (int)((unsigned)NS << 16);            // tensor_dim0[15:0]=NS
  g1[2] = (int)(16u << 16);                     // tensor_dim0 hi=0 | tensor_dim1=16
  g1[3] = (int)(32u << 16);                     // tensor_dim1 hi=0 | tile_dim0=32
  g1[4] = 16;                                   // tile_dim1=16, tile_dim2=0
  g1[5] = NS;                                   // tensor_dim0_stride[31:0]
  g1[6] = 0;                                    // stride hi / dim1_stride (unused)
  g1[7] = 0;
  const v4i z4 = {0, 0, 0, 0};
  const v8i z8 = {0, 0, 0, 0, 0, 0, 0, 0};
  __builtin_amdgcn_tensor_load_to_lds(g0, g1, z4, z4, z8, 0);
}
#elif HAVE_ASYNC_LDS
// Fallback: 4 x async b128 ops, 32 lanes x 16B each.
__device__ __forceinline__ void async_load_dmask(const float* __restrict__ dmr,
                                                 float* buf, int lane) {
  const int rbase = lane >> 3;
  const int cf    = (lane & 7) * 4;
#pragma unroll
  for (int i = 0; i < 4; ++i) {
    const int row = i * 4 + rbase;
    __builtin_amdgcn_global_load_async_to_lds_b128(
        (g4i_t)(dmr + (size_t)row * NS + cf),
        (l4i_t)(buf + row * 32 + cf), 0, 0);
  }
}
#endif

// ---------------------------------------------------------------------------
// Projection: Y = X @ W^T + bias, reshaped to heads.
// One wave computes a 16x16 tile of the (4096 x 1024) output.
// VT==0: store bf16 as (B,H,S,Dh). VT==1: store V transposed (B,H,Dh,S).
// ---------------------------------------------------------------------------
template <int VT>
__global__ __launch_bounds__(256) void proj_heads_kernel(
    const float* __restrict__ X, const float* __restrict__ W,
    const float* __restrict__ bias, bf16_t* __restrict__ Y)
{
  const int wv   = threadIdx.x >> 5;
  const int lane = threadIdx.x & 31;
  const int l    = lane & 15;
  const int half = lane >> 4;
  const int tile = blockIdx.x * 8 + wv;          // 16384 tiles = 256 x 64
  const int m0 = (tile >> 6) << 4;
  const int n0 = (tile & 63) << 4;

  const int o = n0 + l;                          // output feature (col of C)
  v8f acc;
  const float bval = bias[o];
#pragma unroll
  for (int r = 0; r < 8; ++r) acc[r] = bval;

  const float* xrow = X + (size_t)(m0 + l) * NE; // A row for this lane
  const float* wrow = W + (size_t)o * NE;        // B col (contiguous in e)

  for (int kk = 0; kk < NE; kk += 32) {
    // A fragment (16x32 bf16): lane = row m0+l; K = 16*half + j (contiguous)
    v8f xlo = *(const v8f*)(xrow + kk + 16 * half);
    v8f xhi = *(const v8f*)(xrow + kk + 16 * half + 8);
    v16bf a, b;
#pragma unroll
    for (int i = 0; i < 8; ++i) { a[i] = f2bf(xlo[i]); a[i + 8] = f2bf(xhi[i]); }
    // B fragment (32x16): lane = col o; K = 16*half + j (contiguous in W row)
    v16f wv16 = *(const v16f*)(wrow + kk + 16 * half);
#pragma unroll
    for (int i = 0; i < 16; ++i) b[i] = f2bf(wv16[i]);
    acc = wmma_bf16(a, b, acc);
  }

  const int hh = o >> 6;
  const int dd = o & (DH - 1);
#pragma unroll
  for (int r = 0; r < 8; ++r) {
    const int row = m0 + r + 8 * half;           // C layout: VGPR r -> M=r+8h
    const int bb  = row >> 11;                   // / NS
    const int ss  = row & (NS - 1);
    if (VT)
      Y[((size_t)(bb * NH + hh) * DH + dd) * NS + ss] = f2bf(acc[r]);
    else
      Y[((size_t)(bb * NH + hh) * NS + ss) * DH + dd] = f2bf(acc[r]);
  }
}

// ---------------------------------------------------------------------------
// Flash-style attention: one wave owns a 16-query tile, streams 32 keys/iter.
// ---------------------------------------------------------------------------
__global__ __launch_bounds__(256) void attn_kernel(
    const bf16_t* __restrict__ Q, const bf16_t* __restrict__ K,
    const bf16_t* __restrict__ Vt, const int* __restrict__ amask,
    const float* __restrict__ dmask, float* __restrict__ out)
{
  __shared__ __align__(16) bf16_t plds[8][16 * 32]; // 1 KiB / wave (P tile)
#if HAVE_LDS_STAGE
  __shared__ __align__(16) float dbuf[8][2][16 * 32]; // 2x2 KiB / wave (dmask)
#endif
  const int wv   = threadIdx.x >> 5;
  const int lane = threadIdx.x & 31;
  const int l    = lane & 15;
  const int half = lane >> 4;
  const int tile = blockIdx.x * 8 + wv;          // 4096 tiles = 32 bh x 128 q
  const int bh = tile >> 7;
  const int q0 = (tile & 127) << 4;
  const int b  = bh >> 4;

  // Q A-fragments (two K-chunks over Dh=64)
  const bf16_t* qrow = Q + ((size_t)bh * NS + q0 + l) * DH;
  v16bf qa[2];
#pragma unroll
  for (int f = 0; f < 2; ++f) {
    v8bf lo = *(const v8bf*)(qrow + 32 * f + 16 * half);
    v8bf hi = *(const v8bf*)(qrow + 32 * f + 16 * half + 8);
#pragma unroll
    for (int i = 0; i < 8; ++i) { qa[f][i] = lo[i]; qa[f][i + 8] = hi[i]; }
  }

  float mrow[8], zrow[8];
  v8f oacc[4];
#pragma unroll
  for (int r = 0; r < 8; ++r) { mrow[r] = -1.0e30f; zrow[r] = 0.0f; }
#pragma unroll
  for (int dt = 0; dt < 4; ++dt)
#pragma unroll
    for (int r = 0; r < 8; ++r) oacc[dt][r] = 0.0f;

  const int*   am  = amask + b * NS;
  const float* dmr = dmask + ((size_t)bh * NS + q0) * NS;

#if HAVE_TDM
  tdm_load_dmask(dmr, &dbuf[wv][0][0]);            // warm tile 0 via TDM
#elif HAVE_ASYNC_LDS
  async_load_dmask(dmr, &dbuf[wv][0][0], lane);    // warm tile 0
#endif

  for (int j = 0; j < NS / 32; ++j) {
    const int k0 = j * 32;
    // ---- scores: two 16x16 tiles (keys k0..+15 and k0+16..+31) ----
    const bf16_t* krow0 = K + ((size_t)bh * NS + k0 + l) * DH;
    const bf16_t* krow1 = krow0 + 16 * DH;
    v8f s0, s1;
#pragma unroll
    for (int r = 0; r < 8; ++r) { s0[r] = 0.0f; s1[r] = 0.0f; }
#pragma unroll
    for (int f = 0; f < 2; ++f) {
      v16bf kb = *(const v16bf*)(krow0 + 32 * f + 16 * half);  // B: K^T rows
      s0 = wmma_bf16(qa[f], kb, s0);
    }
#pragma unroll
    for (int f = 0; f < 2; ++f) {
      v16bf kb = *(const v16bf*)(krow1 + 32 * f + 16 * half);
      s1 = wmma_bf16(qa[f], kb, s1);
    }

    if (j + 1 < NS / 32) {
      // Warm caches for next chunk: each K row / Vt segment is one line.
      __builtin_prefetch(K + ((size_t)bh * NS + k0 + 32 + lane) * DH, 0, 3);
      __builtin_prefetch(Vt + (size_t)(bh * DH + lane) * NS + k0 + 32, 0, 3);
      __builtin_prefetch(Vt + (size_t)(bh * DH + 32 + lane) * NS + k0 + 32, 0, 3);
    }

#if HAVE_TDM
    // Kick next tile's DMA, then wait for current (TDM ops are in-order:
    // <=1 outstanding leaves only the new one in flight).
    if (j + 1 < NS / 32) {
      tdm_load_dmask(dmr + k0 + 32, &dbuf[wv][(j + 1) & 1][0]);
      WAIT_TENSORCNT(1);
    } else {
      WAIT_TENSORCNT(0);
    }
    asm volatile("" ::: "memory");
    const float* cbuf = &dbuf[wv][j & 1][0];
#elif HAVE_ASYNC_LDS
    if (j + 1 < NS / 32) {
      async_load_dmask(dmr + k0 + 32, &dbuf[wv][(j + 1) & 1][0], lane);
      WAIT_ASYNCCNT(4);
    } else {
      WAIT_ASYNCCNT(0);
    }
    asm volatile("" ::: "memory");
    const float* cbuf = &dbuf[wv][j & 1][0];
#endif

    // ---- key padding mask (per column = per lane) ----
    const bool masked0 = (am[k0 + l] == 0);
    const bool masked1 = (am[k0 + 16 + l] == 0);
#pragma unroll
    for (int r = 0; r < 8; ++r) {
      s0[r] = masked0 ? -1.0e30f : s0[r];
      s1[r] = masked1 ? -1.0e30f : s1[r];
    }

    // ---- online softmax + dropout; stage P tile into LDS as bf16 ----
#pragma unroll
    for (int r = 0; r < 8; ++r) {
      const int row = r + 8 * half;              // C layout row for VGPR r
      float a = s0[r], c = s1[r];
      float tm = fmaxf(a, c);
#pragma unroll
      for (int off = 1; off < 16; off <<= 1)
        tm = fmaxf(tm, __shfl_xor(tm, off, 32)); // reduce within 16-lane half
      const float mn   = fmaxf(mrow[r], tm);
      const float corr = __expf(mrow[r] - mn);
      const float p0 = __expf(a - mn);
      const float p1 = __expf(c - mn);
      float rs = p0 + p1;
#pragma unroll
      for (int off = 1; off < 16; off <<= 1)
        rs += __shfl_xor(rs, off, 32);
      zrow[r] = zrow[r] * corr + rs;
      mrow[r] = mn;
#pragma unroll
      for (int dt = 0; dt < 4; ++dt) oacc[dt][r] = oacc[dt][r] * corr;
#if HAVE_LDS_STAGE
      const float dm0 = cbuf[row * 32 + l];
      const float dm1 = cbuf[row * 32 + 16 + l];
#else
      const float dm0 = dmr[(size_t)row * NS + k0 + l];
      const float dm1 = dmr[(size_t)row * NS + k0 + 16 + l];
#endif
      plds[wv][row * 32 + l]      = f2bf(p0 * dm0);
      plds[wv][row * 32 + 16 + l] = f2bf(p1 * dm1);
    }

    // Same-wave LDS ops are in-order (DScnt); just stop compiler reordering.
    asm volatile("" ::: "memory");

    // ---- reload P as A-fragment (16x32 bf16) ----
    v16bf pa;
    {
      v8bf lo = *(const v8bf*)&plds[wv][l * 32 + 16 * half];
      v8bf hi = *(const v8bf*)&plds[wv][l * 32 + 16 * half + 8];
#pragma unroll
      for (int i = 0; i < 8; ++i) { pa[i] = lo[i]; pa[i + 8] = hi[i]; }
    }

    // ---- O += P @ V (V transposed -> contiguous B-fragment loads) ----
#pragma unroll
    for (int dt = 0; dt < 4; ++dt) {
      v16bf vb = *(const v16bf*)(Vt + (size_t)(bh * DH + dt * 16 + l) * NS +
                                 k0 + 16 * half);
      oacc[dt] = wmma_bf16(pa, vb, oacc[dt]);
    }
  }

  // ---- epilogue: divide by softmax denominator, store f32 (B,H,S,Dh) ----
#pragma unroll
  for (int r = 0; r < 8; ++r) {
    const float z  = zrow[r];
    const float iz = (z > 0.0f) ? (1.0f / z) : 0.0f;
    const int row  = q0 + r + 8 * half;
#pragma unroll
    for (int dt = 0; dt < 4; ++dt)
      out[((size_t)bh * NS + row) * DH + dt * 16 + l] = oacc[dt][r] * iz;
  }
}

// ---------------------------------------------------------------------------
extern "C" void kernel_launch(void* const* d_in, const int* in_sizes, int n_in,
                              void* d_out, int out_size, void* d_ws, size_t ws_size,
                              hipStream_t stream) {
  const float* query = (const float*)d_in[0];
  const float* key   = (const float*)d_in[1];
  const float* value = (const float*)d_in[2];
  const int*   amask = (const int*)d_in[3];
  const float* dmask = (const float*)d_in[4];
  const float* Wq = (const float*)d_in[5];
  const float* bq = (const float*)d_in[6];
  const float* Wk = (const float*)d_in[7];
  const float* bk = (const float*)d_in[8];
  const float* Wv = (const float*)d_in[9];
  const float* bv = (const float*)d_in[10];
  float* out = (float*)d_out;

  bf16_t* qws  = (bf16_t*)d_ws;                  // 8 MiB
  bf16_t* kws  = qws + (size_t)NM * NE;          // 8 MiB
  bf16_t* vtws = kws + (size_t)NM * NE;          // 8 MiB (transposed per head)

  dim3 pb(256), pg(2048);                        // 16384 waves = 256x64 tiles
  proj_heads_kernel<0><<<pg, pb, 0, stream>>>(query, Wq, bq, qws);
  proj_heads_kernel<0><<<pg, pb, 0, stream>>>(key,   Wk, bk, kws);
  proj_heads_kernel<1><<<pg, pb, 0, stream>>>(value, Wv, bv, vtws);

  attn_kernel<<<dim3(512), dim3(256), 0, stream>>>(qws, kws, vtws, amask,
                                                   dmask, out);
}
